// Model_73572789780577
// MI455X (gfx1250) — matrix operand order repaired and checked by reading, not compiled, not run on hardware
//
#include <hip/hip_runtime.h>
#include <math.h>

typedef __attribute__((ext_vector_type(16))) _Float16 v16h;
typedef __attribute__((ext_vector_type(8)))  float    v8f;
typedef __attribute__((ext_vector_type(4)))  unsigned v4u;
typedef __attribute__((ext_vector_type(4)))  int      v4i;
typedef __attribute__((ext_vector_type(4)))  float    v4f;

#define NLVL      10
#define HID       64
#define KPAD      64
#define ROWH      72          // padded row stride (halfs): 144B, 16B-aligned, conflict-reduced
#define TABLE_SZ  65536
#define SELU_S    1.0507009873554805f
#define SELU_A    1.6732632423543772f

union HF { v16h v; v4u q[2]; };

__device__ __forceinline__ float selu_f(float x) {
    return SELU_S * (x > 0.f ? x : SELU_A * (expf(x) - 1.f));
}

// x = A^{-1} b via adjugate (Cramer)
__device__ __forceinline__ void solve3(const float A[3][3], const float b[3], float x[3]) {
    float c00 = A[1][1]*A[2][2] - A[1][2]*A[2][1];
    float c01 = A[1][2]*A[2][0] - A[1][0]*A[2][2];
    float c02 = A[1][0]*A[2][1] - A[1][1]*A[2][0];
    float det = A[0][0]*c00 + A[0][1]*c01 + A[0][2]*c02;
    float inv = 1.0f / det;
    float c10 = A[0][2]*A[2][1] - A[0][1]*A[2][2];
    float c11 = A[0][0]*A[2][2] - A[0][2]*A[2][0];
    float c12 = A[0][1]*A[2][0] - A[0][0]*A[2][1];
    float c20 = A[0][1]*A[1][2] - A[0][2]*A[1][1];
    float c21 = A[0][2]*A[1][0] - A[0][0]*A[1][2];
    float c22 = A[0][0]*A[1][1] - A[0][1]*A[1][0];
    x[0] = (c00*b[0] + c10*b[1] + c20*b[2]) * inv;
    x[1] = (c01*b[0] + c11*b[1] + c21*b[2]) * inv;
    x[2] = (c02*b[0] + c12*b[1] + c22*b[2]) * inv;
}

// A fragment, 16x32 f16, per CDNA5 ISA layout:
// lanes 0-15: K = kb..kb+7 and kb+16..kb+23 (kb = kt*32); lanes 16-31: +8.
__device__ __forceinline__ v16h ldsA(const _Float16* fb, int mt, int kt, int lane) {
    int M  = mt * 16 + (lane & 15);
    int kb = kt * 32 + ((lane & 16) ? 8 : 0);
    const v4u* p = (const v4u*)(fb + M * ROWH + kb);
    HF f; f.q[0] = p[0]; f.q[1] = p[2];
    return f.v;
}
// B fragment, 32x16 f16, weights stored [N][Kpad]: two 8-half chunks per lane.
// lanes 0-15 hold K kb..kb+7 and kb+16..kb+23 (kb = kt*32), lanes 16-31 hold +8.
__device__ __forceinline__ v16h ldsB(const _Float16* wt, int kt, int nt, int lane) {
    int N  = nt * 16 + (lane & 15);
    int kb = kt * 32 + ((lane & 16) ? 8 : 0);
    const v4u* p = (const v4u*)(wt + N * ROWH + kb);
    HF f; f.q[0] = p[0]; f.q[1] = p[2];
    return f.v;
}

__global__ void tet_zero_alpha(float* __restrict__ a, int V) {
    int i = blockIdx.x * 256 + threadIdx.x;
    if (i < V) a[i] = 0.f;
}

__global__ __launch_bounds__(256) void tet_fused_kernel(
    const float* __restrict__ vertices, const int* __restrict__ indices,
    const float* __restrict__ center,   const float* __restrict__ tables,
    const float* __restrict__ W1, const float* __restrict__ b1,
    const float* __restrict__ W2, const float* __restrict__ b2,
    const float* __restrict__ W3, const float* __restrict__ b3,
    float* __restrict__ out_alpha, float* __restrict__ out_vcol,
    int V, int T)
{
    __shared__ __align__(16) _Float16 sW[(HID + HID + 16) * ROWH]; // W1t,W2t,W3t as [N][Kpad]
    __shared__ __align__(16) _Float16 sF[8 * 32 * ROWH];           // per-wave feature/hidden tile
    __shared__ float sB[144];                                      // b1(64) b2(64) b3pad(16)

    const int tid  = threadIdx.x;
    const int lane = tid & 31;
    const int wave = tid >> 5;
    _Float16* W1t = sW;
    _Float16* W2t = sW + HID * ROWH;
    _Float16* W3t = sW + 2 * HID * ROWH;
    _Float16* fb  = sF + wave * 32 * ROWH;

    // ---- stage weights (f32 -> f16, transposed to [N][K]) ----
    for (int i = tid; i < HID * KPAD; i += 256) {
        int n = i >> 6, k = i & 63;
        W1t[n * ROWH + k] = (_Float16)((k < 40) ? W1[k * HID + n] : 0.f);
        W2t[n * ROWH + k] = (_Float16)(W2[k * HID + n]);
    }
    for (int i = tid; i < 16 * KPAD; i += 256) {
        int n = i >> 6, k = i & 63;
        W3t[n * ROWH + k] = (_Float16)((n < 13) ? W3[k * 13 + n] : 0.f);
    }
    for (int i = tid; i < 144; i += 256) {
        float v;
        if (i < 64)       v = b1[i];
        else if (i < 128) v = b2[i - 64];
        else { int n = i - 128; v = (n < 13) ? b3[n] : 0.f; }
        sB[i] = v;
    }

    // ---- per-tet scalar phase (one tet per lane, 32 tets per wave) ----
    int  t      = blockIdx.x * 256 + tid;
    bool active = (t < T);
    int  tc     = active ? t : (T - 1);   // clamp so EXEC stays full for WMMA

    v4i ii = ((const v4i*)indices)[tc];
    int idx[4] = { ii.x, ii.y, ii.z, ii.w };
    float vt[4][3];
#pragma unroll
    for (int j = 0; j < 4; ++j) {
        const float* p = vertices + (size_t)idx[j] * 3;
        vt[j][0] = p[0]; vt[j][1] = p[1]; vt[j][2] = p[2];
    }

    // circumcenter: A c = b, A = 2(v_i - v0) + 1e-6 I
    float A[3][3], bb[3];
    float s0 = vt[0][0]*vt[0][0] + vt[0][1]*vt[0][1] + vt[0][2]*vt[0][2];
#pragma unroll
    for (int i = 0; i < 3; ++i) {
#pragma unroll
        for (int d = 0; d < 3; ++d)
            A[i][d] = 2.f * (vt[i+1][d] - vt[0][d]) + ((i == d) ? 1e-6f : 0.f);
        bb[i] = vt[i+1][0]*vt[i+1][0] + vt[i+1][1]*vt[i+1][1] + vt[i+1][2]*vt[i+1][2] - s0;
    }
    float c[3]; solve3(A, bb, c);
    float rdx = c[0]-vt[0][0], rdy = c[1]-vt[0][1], rdz = c[2]-vt[0][2];
    float radius = sqrtf(rdx*rdx + rdy*rdy + rdz*rdz);

    // project circumcenter into tet
    float E[3][3];
#pragma unroll
    for (int i = 0; i < 3; ++i)
#pragma unroll
        for (int d = 0; d < 3; ++d) E[i][d] = vt[i+1][d] - vt[0][d];
    float M3[3][3], rhs[3];
#pragma unroll
    for (int i = 0; i < 3; ++i) {
#pragma unroll
        for (int j = 0; j < 3; ++j)
            M3[i][j] = E[i][0]*E[j][0] + E[i][1]*E[j][1] + E[i][2]*E[j][2] + ((i==j)?1e-6f:0.f);
        rhs[i] = E[i][0]*(c[0]-vt[0][0]) + E[i][1]*(c[1]-vt[0][1]) + E[i][2]*(c[2]-vt[0][2]);
    }
    float lam[3]; solve3(M3, rhs, lam);
    float bar[4] = { 1.f - lam[0] - lam[1] - lam[2], lam[0], lam[1], lam[2] };
    float bs = 0.f;
#pragma unroll
    for (int j = 0; j < 4; ++j) { bar[j] = fminf(fmaxf(bar[j], 0.f), 1.f); bs += bar[j]; }
    float binv = 1.f / fmaxf(bs, 1e-6f);
    float cc[3] = {0.f, 0.f, 0.f};
#pragma unroll
    for (int j = 0; j < 4; ++j) {
        float w = bar[j] * binv;
        cc[0] += w * vt[j][0]; cc[1] += w * vt[j][1]; cc[2] += w * vt[j][2];
    }

    // contract
    float nx[3] = { (cc[0]-center[0])*0.5f, (cc[1]-center[1])*0.5f, (cc[2]-center[2])*0.5f };
    float nrm = fmaxf(sqrtf(nx[0]*nx[0] + nx[1]*nx[1] + nx[2]*nx[2]), 1e-9f);
    float cf  = (nrm <= 1.f) ? 1.f : (2.f - 1.f/nrm) / nrm;
    float cs  = (radius * 0.5f) * ((nrm <= 1.f) ? 1.f : 1.f/(nrm*nrm));
    float cr  = cs * 0.5f;
    float pos[3];
#pragma unroll
    for (int d = 0; d < 3; ++d) {
        float p = nx[d] * cf * 0.25f + 0.5f;
        pos[d] = fminf(fmaxf(p, 0.f), 0.999999f);
    }

    // hash encode -> f16 features in wave-private LDS tile, row = lane, K padded to 64
    for (int l = 0; l < NLVL; ++l) {
        float res  = (float)(16 << l);
        float scal = erff(1.0f / sqrtf(8.f * (float)l * cr + 1e-12f));
        float w[3]; unsigned fi[3];
#pragma unroll
        for (int d = 0; d < 3; ++d) {
            float xx = pos[d] * res;
            float ff = floorf(xx);
            w[d]  = xx - ff;
            fi[d] = (unsigned)ff;
        }
        float acc0 = 0.f, acc1 = 0.f, acc2 = 0.f, acc3 = 0.f;
#pragma unroll
        for (int cz = 0; cz < 2; ++cz)
#pragma unroll
        for (int cy = 0; cy < 2; ++cy)
#pragma unroll
        for (int cx = 0; cx < 2; ++cx) {
            unsigned h = (fi[0] + (unsigned)cx)
                       ^ ((fi[1] + (unsigned)cy) * 2654435761u)
                       ^ ((fi[2] + (unsigned)cz) * 805459861u);
            h &= (unsigned)(TABLE_SZ - 1);
            float wt = (cx ? w[0] : 1.f - w[0]) * (cy ? w[1] : 1.f - w[1]) * (cz ? w[2] : 1.f - w[2]);
            v4f tv = ((const v4f*)tables)[(size_t)l * TABLE_SZ + h];
            acc0 += tv.x * wt; acc1 += tv.y * wt; acc2 += tv.z * wt; acc3 += tv.w * wt;
        }
        _Float16* fr = fb + lane * ROWH + l * 4;
        fr[0] = (_Float16)(acc0 * scal); fr[1] = (_Float16)(acc1 * scal);
        fr[2] = (_Float16)(acc2 * scal); fr[3] = (_Float16)(acc3 * scal);
    }
#pragma unroll
    for (int k = 40; k < 64; ++k) fb[lane * ROWH + k] = (_Float16)0.f;

    // max edge length
    float el = 0.f;
#pragma unroll
    for (int a0 = 0; a0 < 3; ++a0)
#pragma unroll
        for (int b0 = a0 + 1; b0 < 4; ++b0) {
            float dx = vt[a0][0]-vt[b0][0], dy = vt[a0][1]-vt[b0][1], dz = vt[a0][2]-vt[b0][2];
            el = fmaxf(el, sqrtf(dx*dx + dy*dy + dz*dz));
        }

    __syncthreads();

    const int nn0 = lane & 15;

    // ---- layer 1: [32 x 64] @ [64 x 64] via v_wmma_f32_16x16x32_f16 ----
    v8f acc[2][4];
#pragma unroll
    for (int mt = 0; mt < 2; ++mt)
#pragma unroll
    for (int nt = 0; nt < 4; ++nt) {
        v8f a = {};
#pragma unroll
        for (int kt = 0; kt < 2; ++kt) {
            v16h av = ldsA(fb,  mt, kt, lane);
            v16h bv = ldsB(W1t, kt, nt, lane);
            a = __builtin_amdgcn_wmma_f32_16x16x32_f16(false, av, false, bv, (short)0, a, false, false);
        }
        acc[mt][nt] = a;
    }
    __syncthreads();
#pragma unroll
    for (int mt = 0; mt < 2; ++mt)
#pragma unroll
    for (int nt = 0; nt < 4; ++nt) {
        float bias = sB[nt * 16 + nn0];
#pragma unroll
        for (int r = 0; r < 8; ++r) {
            int Mrow = mt * 16 + ((lane & 16) ? 8 + r : r);
            fb[Mrow * ROWH + nt * 16 + nn0] = (_Float16)selu_f(acc[mt][nt][r] + bias);
        }
    }
    __syncthreads();

    // ---- layer 2: [32 x 64] @ [64 x 64] ----
#pragma unroll
    for (int mt = 0; mt < 2; ++mt)
#pragma unroll
    for (int nt = 0; nt < 4; ++nt) {
        v8f a = {};
#pragma unroll
        for (int kt = 0; kt < 2; ++kt) {
            v16h av = ldsA(fb,  mt, kt, lane);
            v16h bv = ldsB(W2t, kt, nt, lane);
            a = __builtin_amdgcn_wmma_f32_16x16x32_f16(false, av, false, bv, (short)0, a, false, false);
        }
        acc[mt][nt] = a;
    }
    __syncthreads();
#pragma unroll
    for (int mt = 0; mt < 2; ++mt)
#pragma unroll
    for (int nt = 0; nt < 4; ++nt) {
        float bias = sB[64 + nt * 16 + nn0];
#pragma unroll
        for (int r = 0; r < 8; ++r) {
            int Mrow = mt * 16 + ((lane & 16) ? 8 + r : r);
            fb[Mrow * ROWH + nt * 16 + nn0] = (_Float16)selu_f(acc[mt][nt][r] + bias);
        }
    }
    __syncthreads();

    // ---- layer 3: [32 x 64] @ [64 x 16(pad of 13)] ----
    v8f acc3v[2];
#pragma unroll
    for (int mt = 0; mt < 2; ++mt) {
        v8f a = {};
#pragma unroll
        for (int kt = 0; kt < 2; ++kt) {
            v16h av = ldsA(fb,  mt, kt, lane);
            v16h bv = ldsB(W3t, kt, 0, lane);
            a = __builtin_amdgcn_wmma_f32_16x16x32_f16(false, av, false, bv, (short)0, a, false, false);
        }
        acc3v[mt] = a;
    }
    __syncthreads();

    // distribute outputs back to per-tet lanes via wave tile (as f32, row stride 18 -> conflict-free)
    float* ob = (float*)fb;
    {
        float b3v = sB[128 + nn0];
#pragma unroll
        for (int mt = 0; mt < 2; ++mt)
#pragma unroll
        for (int r = 0; r < 8; ++r) {
            int Mrow = mt * 16 + ((lane & 16) ? 8 + r : r);
            ob[Mrow * 18 + nn0] = acc3v[mt][r] + b3v;
        }
    }
    __syncthreads();

    float o[13];
#pragma unroll
    for (int j = 0; j < 13; ++j) o[j] = ob[lane * 18 + j];

    // ---- finalize ----
    float d0 = fminf(fmaxf(o[0] - 1.0f, -30.f), 10.f);
    float density = expf(d0);
    float alpha = 1.f - expf(-density * el);

    if (active) {
        unsigned au = __float_as_uint(alpha);   // alpha >= 0: uint compare == float compare
#pragma unroll
        for (int j = 0; j < 4; ++j)
            atomicMax((unsigned*)out_alpha + idx[j], au);

        float bse[3] = { o[1] + 0.5f, o[2] + 0.5f, o[3] + 0.5f };
        float g[3][3];
#pragma unroll
        for (int ci = 0; ci < 3; ++ci)
#pragma unroll
            for (int d = 0; d < 3; ++d)
                g[ci][d] = fmaxf(bse[ci], 0.f) * tanhf(o[4 + ci * 3 + d]);

        float* vo = out_vcol + (size_t)t * 12;
#pragma unroll
        for (int vx = 0; vx < 4; ++vx) {
            float dx = vt[vx][0] - cc[0], dy = vt[vx][1] - cc[1], dz = vt[vx][2] - cc[2];
            float inv = 1.0f / sqrtf(dx*dx + dy*dy + dz*dz + 1e-8f);
            float ox = dx * inv, oy = dy * inv, oz = dz * inv;
#pragma unroll
            for (int ci = 0; ci < 3; ++ci)
                vo[vx * 3 + ci] = bse[ci] + g[ci][0]*ox + g[ci][1]*oy + g[ci][2]*oz;
        }
    }
}

extern "C" void kernel_launch(void* const* d_in, const int* in_sizes, int n_in,
                              void* d_out, int out_size, void* d_ws, size_t ws_size,
                              hipStream_t stream) {
    const float* vertices = (const float*)d_in[0];
    const int*   indices  = (const int*)  d_in[1];
    const float* center   = (const float*)d_in[2];
    const float* tables   = (const float*)d_in[3];
    const float* W1 = (const float*)d_in[4];
    const float* b1 = (const float*)d_in[5];
    const float* W2 = (const float*)d_in[6];
    const float* b2 = (const float*)d_in[7];
    const float* W3 = (const float*)d_in[8];
    const float* b3 = (const float*)d_in[9];

    int V = in_sizes[0] / 3;
    int T = in_sizes[1] / 4;

    float* out_alpha = (float*)d_out;
    float* out_vcol  = out_alpha + V;

    hipLaunchKernelGGL(tet_zero_alpha, dim3((V + 255) / 256), dim3(256), 0, stream, out_alpha, V);
    hipLaunchKernelGGL(tet_fused_kernel, dim3((T + 255) / 256), dim3(256), 0, stream,
                       vertices, indices, center, tables, W1, b1, W2, b2, W3, b3,
                       out_alpha, out_vcol, V, T);
}
